// LstmCnnFeatGate_56839597195268
// MI455X (gfx1250) — matrix operand order repaired and checked by reading, not compiled, not run on hardware
//
#include <hip/hip_runtime.h>
#include <hip/hip_bf16.h>

typedef __attribute__((ext_vector_type(16))) _Float16 v16h;
typedef __attribute__((ext_vector_type(8)))  _Float16 v8h;
typedef __attribute__((ext_vector_type(8)))  float    v8f;

#define LDSP __attribute__((address_space(3)))

#define GF_BIAS    1
#define GF_ACC     2
#define GF_TANH    4
#define GF_SIGMOID 8

#define Bsz   32
#define Ssz   128
#define BS    4096      // B*S
#define Wch   20
#define CEm   50
#define NFm   64
#define TOTF  256
#define CFm   300
#define WDm   300
#define Hm    512
#define G4    2048
#define KP300 320
#define NP300 304
#define NEGV  (-10000.0f)

__device__ __forceinline__ float sigf(float x) { return 1.0f / (1.0f + __expf(-x)); }

// ---------------------------------------------------------------- embeddings
__global__ void k_gather_word(const int* __restrict__ ids,
                              const float* __restrict__ emb,
                              float* __restrict__ out) {
  int i = blockIdx.x * blockDim.x + threadIdx.x;
  if (i >= BS * WDm) return;
  int bs = i / WDm, d = i - bs * WDm;
  out[i] = emb[(size_t)ids[bs] * WDm + d];
}

// ---------------------------------------------------------------- char CNN
__global__ void k_char_cnn(const int* __restrict__ char_ids,
                           const float* __restrict__ char_embed,
                           const float* __restrict__ w2, const float* __restrict__ b2,
                           const float* __restrict__ w3, const float* __restrict__ b3,
                           const float* __restrict__ w4, const float* __restrict__ b4,
                           const float* __restrict__ w5, const float* __restrict__ b5,
                           float* __restrict__ ch) {
  __shared__ float x[Wch * CEm];
  const int word = blockIdx.x;
  const int tid = threadIdx.x;
  for (int i = tid; i < Wch * CEm; i += blockDim.x) {
    int c = i / CEm, d = i - c * CEm;
    x[i] = char_embed[(size_t)char_ids[word * Wch + c] * CEm + d];
  }
  __syncthreads();
  const int f = tid;            // 0..255
  const int g = f >> 6;         // kernel-size group
  const int j = f & 63;         // filter
  const int k = g + 2;
  const float* w = (g == 0) ? w2 : (g == 1) ? w3 : (g == 2) ? w4 : w5;
  const float bias = ((g == 0) ? b2 : (g == 1) ? b3 : (g == 2) ? b4 : b5)[j];
  float best = 0.0f;  // max(relu(y)) >= 0
  for (int pos = 0; pos <= Wch - k; ++pos) {
    float acc = bias;
    for (int p = 0; p < k; ++p)
      for (int d = 0; d < CEm; ++d)
        acc += x[(pos + p) * CEm + d] * w[((p * CEm) + d) * NFm + j];
    best = fmaxf(best, acc);
  }
  ch[(size_t)word * TOTF + f] = best;
}

// ----------------------------------------------------- f32 -> f16 (zero pad)
__global__ void k_cvt_pad(const float* __restrict__ src, _Float16* __restrict__ dst,
                          int R, int C, int Rp, int Cp) {
  int i = blockIdx.x * blockDim.x + threadIdx.x;
  if (i >= Rp * Cp) return;
  int r = i / Cp, c = i - r * Cp;
  float v = (r < R && c < C) ? src[(size_t)r * C + c] : 0.0f;
  dst[i] = (_Float16)v;
}

// ------------------------------------------------ WMMA GEMM: C = A*Wt^T (+b)
// A: M x Kp f16 row-major (Kp % 32 == 0, Kp <= KP300). Wt: Npad x Kp f16
// row-major (Npad >= 16*ceil(N/16), zero padded). C: M x N f32.
// The block's shared 16 x Kp A tile is staged into LDS with async copies.
__global__ void k_gemm(const _Float16* __restrict__ A, const _Float16* __restrict__ Wt,
                       const float* __restrict__ bias, float* __restrict__ C,
                       int M, int N, int Kp, int flags) {
  __shared__ __align__(16) _Float16 atile[16 * KP300];   // 10 KB max
  const int lane  = threadIdx.x & 31;
  const int wave  = threadIdx.x >> 5;
  const int ntiles = (N + 15) >> 4;
  const int ntile = blockIdx.x * (blockDim.x >> 5) + wave;
  const int mtile = blockIdx.y;
  const int row0  = mtile << 4;

  // ---- async-stage the contiguous 16 x Kp A tile into LDS (all threads) ----
  {
    const _Float16* asrc = A + (size_t)row0 * Kp;
    LDSP _Float16* adst = (LDSP _Float16*)&atile[0];
    const int chunks = 2 * Kp;                 // 16-byte chunks
    for (int i = threadIdx.x; i < chunks; i += blockDim.x) {
      unsigned dst = (unsigned)(unsigned long long)(adst + i * 8);
      const _Float16* src = asrc + (size_t)i * 8;
      asm volatile("global_load_async_to_lds_b128 %0, %1, off"
                   :: "v"(dst), "v"(src) : "memory");
    }
    asm volatile("s_wait_asynccnt 0x0" ::: "memory");
  }
  __syncthreads();
  if (ntile >= ntiles) return;           // wave-uniform exit, EXEC stays all-1s

  const int col0 = ntile << 4;
  const int hi   = lane >> 4;
  const int wrow = col0 + (lane & 15);
  const int colc = col0 + (lane & 15);
  v8f acc = {};
  if (flags & GF_ACC) {
#pragma unroll
    for (int r = 0; r < 8; ++r) {
      int row = row0 + r + hi * 8;
      acc[r] = (colc < N) ? C[(size_t)row * N + colc] : 0.0f;
    }
  }
  for (int kb = 0; kb < Kp; kb += 32) {
    const _Float16* ap = &atile[(lane & 15) * Kp + kb + hi * 16];
    v8h a0 = *(const v8h*)(ap);
    v8h a1 = *(const v8h*)(ap + 8);
    v16h a;
#pragma unroll
    for (int e = 0; e < 8; ++e) { a[e] = a0[e]; a[e + 8] = a1[e]; }
    const v16h bfr = *(const v16h*)(Wt + (size_t)wrow * Kp + kb + hi * 16);
    acc = __builtin_amdgcn_wmma_f32_16x16x32_f16(false, a, false, bfr,
                                                 (short)0, acc, false, false);
  }
  float bv = 0.0f;
  if ((flags & GF_BIAS) && colc < N) bv = bias[colc];
#pragma unroll
  for (int r = 0; r < 8; ++r) {
    int row = row0 + r + hi * 8;
    if (colc < N) {
      float v = acc[r] + bv;
      if (flags & GF_TANH)    v = tanhf(v);
      if (flags & GF_SIGMOID) v = sigf(v);
      C[(size_t)row * N + colc] = v;
    }
  }
}

// ---------------------------------------------------------------- gate blend
__global__ void k_feats(const float* __restrict__ g, const float* __restrict__ wi,
                        const float* __restrict__ ci, float* __restrict__ f) {
  int i = blockIdx.x * blockDim.x + threadIdx.x;
  if (i >= BS * WDm) return;
  float gv = g[i];
  f[i] = gv * wi[i] + (1.0f - gv) * ci[i];
}

// ------------------------------------------------------- persistent BiLSTM
// blockIdx.x = direction (0 fwd, 1 bwd). 1024 threads = 32 waves.
// Gate GEMM (32 x 2048, K=512) per step via WMMA; h (f16) and the gate
// staging buffer both live in the 320 KB CDNA5 LDS (288 KB total), so the
// serial per-step path never leaves the WGP.
__global__ __launch_bounds__(1024) void k_lstm(
    const float* __restrict__ xg_f, const float* __restrict__ xg_b,
    const _Float16* __restrict__ whh_f16, const _Float16* __restrict__ whh_b16,
    const int* __restrict__ lens,
    float* __restrict__ hout_f, float* __restrict__ hout_b) {
  const int dir = blockIdx.x;
  const float*    xg   = dir ? xg_b : xg_f;
  const _Float16* whh  = dir ? whh_b16 : whh_f16;
  float*          hout = dir ? hout_b : hout_f;

  __shared__ __align__(16) _Float16 hsh[Bsz * Hm];  // 32 KB
  __shared__ float gsh[Bsz * G4];                   // 256 KB

  const int tid  = threadIdx.x;
  const int lane = tid & 31;
  const int wave = tid >> 5;
  for (int i = tid; i < Bsz * Hm; i += blockDim.x) hsh[i] = (_Float16)0.0f;

  const int b = tid & 31;
  const int ubase = (tid >> 5) * 16;
  float creg[16], hreg[16];
#pragma unroll
  for (int u = 0; u < 16; ++u) { creg[u] = 0.0f; hreg[u] = 0.0f; }
  const int myLen = lens[b];
  __syncthreads();

  for (int step = 0; step < Ssz; ++step) {
    const int t = dir ? (Ssz - 1 - step) : step;
    // gates = h @ whh^T + xg[:, t, :]
    for (int i = 0; i < 8; ++i) {
      int tile  = wave + 32 * i;      // 0..255
      int mtile = tile >> 7;          // 0..1
      int ntile = tile & 127;         // 0..127
      int row0 = mtile << 4, col0 = ntile << 4;
      int m  = row0 + (lane & 15);
      int hi = lane >> 4;
      int wrow = col0 + (lane & 15);
      v8f acc = {};
      for (int kb = 0; kb < Hm; kb += 32) {
        const _Float16* ap = &hsh[m * Hm + kb + hi * 16];
        v8h a0 = *(const v8h*)(ap);
        v8h a1 = *(const v8h*)(ap + 8);
        v16h a;
#pragma unroll
        for (int e = 0; e < 8; ++e) { a[e] = a0[e]; a[e + 8] = a1[e]; }
        const v16h bfr = *(const v16h*)(whh + (size_t)wrow * Hm + kb + hi * 16);
        acc = __builtin_amdgcn_wmma_f32_16x16x32_f16(false, a, false, bfr,
                                                     (short)0, acc, false, false);
      }
#pragma unroll
      for (int r = 0; r < 8; ++r) {
        int row = row0 + r + hi * 8;    // batch
        int col = col0 + (lane & 15);   // gate unit
        gsh[row * G4 + col] = acc[r] + xg[((size_t)row * Ssz + t) * G4 + col];
      }
    }
    __syncthreads();
    // pointwise LSTM cell for owned (b, ubase..ubase+15)
    const bool msk = (t < myLen);
#pragma unroll
    for (int u = 0; u < 16; ++u) {
      int hu = ubase + u;
      float gi = gsh[b * G4 + hu];
      float gf = gsh[b * G4 + Hm + hu];
      float gg = gsh[b * G4 + 2 * Hm + hu];
      float go = gsh[b * G4 + 3 * Hm + hu];
      float cn = sigf(gf) * creg[u] + sigf(gi) * tanhf(gg);
      float hn = sigf(go) * tanhf(cn);
      float hnew = msk ? hn : hreg[u];
      creg[u] = msk ? cn : creg[u];
      hreg[u] = hnew;
      hout[((size_t)b * Ssz + t) * Hm + hu] = msk ? hn : 0.0f;
      hsh[b * Hm + hu] = (_Float16)hnew;
    }
    __syncthreads();
  }
}

// ------------------------------------------------------- output projection
__global__ void k_outproj(const float* __restrict__ hf, const float* __restrict__ hb,
                          const float* __restrict__ ow, const float* __restrict__ ob,
                          float* __restrict__ logits) {
  int idx = blockIdx.x * blockDim.x + threadIdx.x;
  if (idx >= BS * 16) return;
  int bs = idx >> 4, l = idx & 15;
  if (l >= 11) return;
  float v;
  if (l < 9) {
    float s = ob[l];
    const float* hfp = hf + (size_t)bs * Hm;
    const float* hbp = hb + (size_t)bs * Hm;
    const float* w   = ow + (size_t)l * (2 * Hm);
    for (int k = 0; k < Hm; ++k) s += hfp[k] * w[k];
    for (int k = 0; k < Hm; ++k) s += hbp[k] * w[Hm + k];
    v = s;
  } else {
    v = NEGV;
  }
  logits[(size_t)bs * 11 + l] = v;
}

// ----------------------------------------------------------- CRF forward
__global__ void k_crf_norm(const float* __restrict__ logits, const int* __restrict__ lens,
                           const float* __restrict__ trn, float* __restrict__ norm) {
  __shared__ float alpha[2][Bsz][11];
  int tid = threadIdx.x;
  int b = tid / 11, j = tid - b * 11;
  bool act = (tid < Bsz * 11);
  if (act) alpha[0][b][j] = logits[(size_t)b * Ssz * 11 + j] + trn[j * 11 + 9];
  __syncthreads();
  int cur = 0;
  for (int t = 1; t < Ssz; ++t) {
    if (act) {
      float mx = -1e30f;
#pragma unroll
      for (int k = 0; k < 11; ++k) mx = fmaxf(mx, alpha[cur][b][k] + trn[j * 11 + k]);
      float s = 0.0f;
#pragma unroll
      for (int k = 0; k < 11; ++k) s += __expf(alpha[cur][b][k] + trn[j * 11 + k] - mx);
      float nw = mx + __logf(s) + logits[((size_t)b * Ssz + t) * 11 + j];
      alpha[cur ^ 1][b][j] = (t < lens[b]) ? nw : alpha[cur][b][j];
    }
    __syncthreads();
    cur ^= 1;
  }
  if (act && j == 0) {
    float mx = -1e30f;
    for (int k = 0; k < 11; ++k) mx = fmaxf(mx, alpha[cur][b][k] + trn[10 * 11 + k]);
    float s = 0.0f;
    for (int k = 0; k < 11; ++k) s += __expf(alpha[cur][b][k] + trn[10 * 11 + k] - mx);
    norm[b] = mx + __logf(s);
  }
}

__global__ void k_crf_gold(const float* __restrict__ logits, const int* __restrict__ labels,
                           const int* __restrict__ lens, const float* __restrict__ trn,
                           const float* __restrict__ norm, float* __restrict__ loglik) {
  int b = threadIdx.x;
  if (b >= Bsz) return;
  int len = lens[b];
  int prev = labels[b * Ssz];
  float s = logits[(size_t)b * Ssz * 11 + prev];     // t=0 emit (len >= 64)
  for (int t = 1; t < Ssz; ++t) {
    int lab = labels[b * Ssz + t];
    if (t < len) {
      s += logits[((size_t)b * Ssz + t) * 11 + lab];
      s += trn[lab * 11 + prev];
    }
    prev = lab;
  }
  s += trn[labels[b * Ssz] * 11 + 9];                // trn[lab0, START]
  s += trn[10 * 11 + labels[b * Ssz + (len - 1)]];   // trn[END, last]
  loglik[b] = s - norm[b];
}

// ============================================================== host launch
extern "C" void kernel_launch(void* const* d_in, const int* in_sizes, int n_in,
                              void* d_out, int out_size, void* d_ws, size_t ws_size,
                              hipStream_t stream) {
  (void)in_sizes; (void)n_in; (void)out_size; (void)ws_size;
  const int*   token_ids  = (const int*)d_in[0];
  const int*   char_ids   = (const int*)d_in[1];
  const int*   lens       = (const int*)d_in[2];
  const int*   labels     = (const int*)d_in[3];
  const float* word_embed = (const float*)d_in[4];
  const float* char_embed = (const float*)d_in[5];
  const float* conv_w2 = (const float*)d_in[6];  const float* conv_b2 = (const float*)d_in[7];
  const float* conv_w3 = (const float*)d_in[8];  const float* conv_b3 = (const float*)d_in[9];
  const float* conv_w4 = (const float*)d_in[10]; const float* conv_b4 = (const float*)d_in[11];
  const float* conv_w5 = (const float*)d_in[12]; const float* conv_b5 = (const float*)d_in[13];
  const float* ff_w    = (const float*)d_in[14]; const float* ff_b    = (const float*)d_in[15];
  const float* wgate_w = (const float*)d_in[16];
  const float* cgate_w = (const float*)d_in[17];
  const float* gate_w  = (const float*)d_in[18];
  const float* wih_f   = (const float*)d_in[19];
  const float* whh_f   = (const float*)d_in[20];
  const float* b_f     = (const float*)d_in[21];
  const float* wih_b   = (const float*)d_in[22];
  const float* whh_b   = (const float*)d_in[23];
  const float* b_b     = (const float*)d_in[24];
  const float* out_w   = (const float*)d_in[25];
  const float* out_b   = (const float*)d_in[26];
  const float* trn     = (const float*)d_in[27];

  float* loglik = (float*)d_out;          // 32
  float* logits = (float*)d_out + Bsz;    // 32*128*11

  // workspace layout
  char* base = (char*)d_ws;
  size_t off = 0;
  auto alloc = [&](size_t bytes) { size_t o = off; off = (off + bytes + 255) & ~(size_t)255; return o; };
  float*     word_in = (float*)(base + alloc((size_t)BS * WDm * 4));
  float*     ch      = (float*)(base + alloc((size_t)BS * TOTF * 4));
  float*     char_in = (float*)(base + alloc((size_t)BS * CFm * 4));
  float*     t1      = (float*)(base + alloc((size_t)BS * WDm * 4));   // tanh arg, then g
  float*     featsb  = (float*)(base + alloc((size_t)BS * WDm * 4));
  float*     xg_f    = (float*)(base + alloc((size_t)BS * G4 * 4));
  float*     xg_b    = (float*)(base + alloc((size_t)BS * G4 * 4));
  float*     h_f     = (float*)(base + alloc((size_t)BS * Hm * 4));
  float*     h_b     = (float*)(base + alloc((size_t)BS * Hm * 4));
  float*     normb   = (float*)(base + alloc((size_t)Bsz * 4));
  _Float16*  act16   = (_Float16*)(base + alloc((size_t)BS * KP300 * 2));  // shared activation buf
  _Float16*  ffw16   = (_Float16*)(base + alloc((size_t)NP300 * TOTF * 2));
  _Float16*  wgw16   = (_Float16*)(base + alloc((size_t)NP300 * KP300 * 2));
  _Float16*  cgw16   = (_Float16*)(base + alloc((size_t)NP300 * KP300 * 2));
  _Float16*  gw16    = (_Float16*)(base + alloc((size_t)NP300 * KP300 * 2));
  _Float16*  wihf16  = (_Float16*)(base + alloc((size_t)G4 * KP300 * 2));
  _Float16*  wihb16  = (_Float16*)(base + alloc((size_t)G4 * KP300 * 2));
  _Float16*  whhf16  = (_Float16*)(base + alloc((size_t)G4 * Hm * 2));
  _Float16*  whhb16  = (_Float16*)(base + alloc((size_t)G4 * Hm * 2));

  auto cvt = [&](const float* s, _Float16* d, int R, int C, int Rp, int Cp) {
    int n = Rp * Cp;
    k_cvt_pad<<<(n + 255) / 256, 256, 0, stream>>>(s, d, R, C, Rp, Cp);
  };
  auto gemm = [&](const _Float16* A, const _Float16* Wt, const float* bias, float* C,
                  int M, int N, int Kp, int flags) {
    int ntiles = (N + 15) / 16;
    dim3 grid((ntiles + 3) / 4, M / 16);
    k_gemm<<<grid, 128, 0, stream>>>(A, Wt, bias, C, M, N, Kp, flags);
  };

  // 1. embeddings + char CNN
  {
    int n = BS * WDm;
    k_gather_word<<<(n + 255) / 256, 256, 0, stream>>>(token_ids, word_embed, word_in);
  }
  k_char_cnn<<<BS, 256, 0, stream>>>(char_ids, char_embed, conv_w2, conv_b2, conv_w3, conv_b3,
                                     conv_w4, conv_b4, conv_w5, conv_b5, ch);

  // 2. weight conversions to f16 (padded)
  cvt(ff_w,    ffw16,  CFm, TOTF, NP300, TOTF);
  cvt(wgate_w, wgw16,  WDm, WDm,  NP300, KP300);
  cvt(cgate_w, cgw16,  CFm, CFm,  NP300, KP300);
  cvt(gate_w,  gw16,   WDm, WDm,  NP300, KP300);
  cvt(wih_f,   wihf16, G4,  WDm,  G4,    KP300);
  cvt(wih_b,   wihb16, G4,  WDm,  G4,    KP300);
  cvt(whh_f,   whhf16, G4,  Hm,   G4,    Hm);
  cvt(whh_b,   whhb16, G4,  Hm,   G4,    Hm);

  // 3. char_in = ch @ ff_w^T + ff_b
  cvt(ch, act16, BS, TOTF, BS, TOTF);
  gemm(act16, ffw16, ff_b, char_in, BS, CFm, TOTF, GF_BIAS);

  // 4. t1 = tanh(word_in @ wgate^T + char_in @ cgate^T)
  cvt(word_in, act16, BS, WDm, BS, KP300);
  gemm(act16, wgw16, nullptr, t1, BS, WDm, KP300, 0);
  cvt(char_in, act16, BS, CFm, BS, KP300);
  gemm(act16, cgw16, nullptr, t1, BS, WDm, KP300, GF_ACC | GF_TANH);

  // 5. g = sigmoid(t1 @ gate^T), feats = g*word_in + (1-g)*char_in
  cvt(t1, act16, BS, WDm, BS, KP300);
  gemm(act16, gw16, nullptr, t1, BS, WDm, KP300, GF_SIGMOID);   // overwrite t1 with g
  {
    int n = BS * WDm;
    k_feats<<<(n + 255) / 256, 256, 0, stream>>>(t1, word_in, char_in, featsb);
  }

  // 6. xg = feats @ wih^T + b  (both directions)
  cvt(featsb, act16, BS, WDm, BS, KP300);
  gemm(act16, wihf16, b_f, xg_f, BS, G4, KP300, GF_BIAS);
  gemm(act16, wihb16, b_b, xg_b, BS, G4, KP300, GF_BIAS);

  // 7. recurrent BiLSTM (persistent block per direction)
  k_lstm<<<2, 1024, 0, stream>>>(xg_f, xg_b, whhf16, whhb16, lens, h_f, h_b);

  // 8. logits + CRF
  k_outproj<<<(BS * 16 + 255) / 256, 256, 0, stream>>>(h_f, h_b, out_w, out_b, logits);
  k_crf_norm<<<1, 352, 0, stream>>>(logits, lens, trn, normb);
  k_crf_gold<<<1, 32, 0, stream>>>(logits, labels, lens, trn, normb, loglik);
}
